// LAtAttrRobertaSelfAttention_137438954127
// MI455X (gfx1250) — matrix operand order repaired and checked by reading, not compiled, not run on hardware
//
#include <hip/hip_runtime.h>
#include <hip/hip_bf16.h>

#define LOG2E 1.4426950408889634f

typedef __attribute__((ext_vector_type(16))) __bf16 v16bf;
typedef __attribute__((ext_vector_type(8)))  __bf16 v8bf;
typedef __attribute__((ext_vector_type(4)))  __bf16 v4bf;
typedef __attribute__((ext_vector_type(8)))  float  v8f;
typedef __attribute__((ext_vector_type(4)))  float  v4f;

static __device__ __forceinline__ __bf16 f2bf(float f) {
  return static_cast<__bf16>(f);    // native bf16 convert on gfx1250 (RNE)
}

static __device__ __forceinline__ v16bf cat16(v8bf lo, v8bf hi) {
  v16bf r;
#pragma unroll
  for (int i = 0; i < 8; ++i) { r[i] = lo[i]; r[i + 8] = hi[i]; }
  return r;
}

// Async global->LDS 16B copy (CDNA5 Tensor/Async path, ASYNCcnt-tracked).
static __device__ __forceinline__ void async_b128(unsigned lds_off, const void* gaddr) {
  asm volatile("global_load_async_to_lds_b128 %0, %1, off"
               :: "v"(lds_off), "v"(gaddr) : "memory");
}
static __device__ __forceinline__ void wait_async0() {
  asm volatile("s_wait_asynccnt 0x0" ::: "memory");
}

// -----------------------------------------------------------------------------
// Kernel 1: QKV projection.  out = hs @ W.T + b   (NT GEMM, M=4096,N=1024,K=1024)
//   z==0 -> Q  stored [B,H,S,D] bf16, pre-scaled by (1/sqrt(D))*log2(e)
//   z==1 -> K  stored [B,H,S,D] bf16
//   z==2 -> V  stored transposed [B,H,D,S] bf16 (so PV B-fragments are contiguous)
// -----------------------------------------------------------------------------
__global__ __launch_bounds__(128) void qkv_kernel(
    const float* __restrict__ hs,
    const float* __restrict__ Wq, const float* __restrict__ bq,
    const float* __restrict__ Wk, const float* __restrict__ bk,
    const float* __restrict__ Wv, const float* __restrict__ bv,
    __bf16* __restrict__ qws, __bf16* __restrict__ kws, __bf16* __restrict__ vtws)
{
  const int mt = blockIdx.x;          // 64 M-tiles of 64 rows
  const int nt = blockIdx.y;          // 16 N-tiles of 64 cols
  const int z  = blockIdx.z;          // 0:Q 1:K 2:V
  const float* W    = (z == 0) ? Wq : ((z == 1) ? Wk : Wv);
  const float* bias = (z == 0) ? bq : ((z == 1) ? bk : bv);

  __shared__ __align__(16) __bf16 lA[64][32];
  __shared__ __align__(16) __bf16 lB[64][32];

  const int tid  = threadIdx.x;
  const int lane = tid & 31;
  const int w    = tid >> 5;          // wave 0..3
  const int hi   = lane >> 4;         // lane-half
  const int lo16 = lane & 15;
  const int mo   = (w & 1) * 32;      // wave sub-tile 32x32
  const int no   = (w >> 1) * 32;

  const int Mbase = mt * 64;
  const int Nbase = nt * 64;

  // vectorized staging map: 4 consecutive floats per thread per step
  const int sr = tid >> 3;            // 0..15 (row group base)
  const int sc4 = (tid & 7) * 4;      // col 0,4,..,28

  v8f acc[2][2];
#pragma unroll
  for (int mi = 0; mi < 2; ++mi)
#pragma unroll
    for (int ni = 0; ni < 2; ++ni) {
      float bv_ = bias[Nbase + no + ni * 16 + lo16]; // bias broadcast per output column
#pragma unroll
      for (int j = 0; j < 8; ++j) acc[mi][ni][j] = bv_;
    }

  for (int kt = 0; kt < 1024; kt += 32) {
    __syncthreads();
#pragma unroll
    for (int i = 0; i < 4; ++i) {     // 64 rows staged in 4 steps of 16 rows
      int r = sr + i * 16;
      v4f a = *(const v4f*)&hs[(size_t)(Mbase + r) * 1024 + kt + sc4];
      v4f bv4 = *(const v4f*)&W [(size_t)(Nbase + r) * 1024 + kt + sc4];
      v4bf ab, bb;
#pragma unroll
      for (int j = 0; j < 4; ++j) { ab[j] = f2bf(a[j]); bb[j] = f2bf(bv4[j]); }
      *(v4bf*)&lA[r][sc4] = ab;
      *(v4bf*)&lB[r][sc4] = bb;
    }
    __syncthreads();

    v16bf bf_[2];
#pragma unroll
    for (int ni = 0; ni < 2; ++ni) {  // B 32x16 frag: lane-half picks K 0-15 / 16-31
      const __bf16* p = &lB[no + ni * 16 + lo16][hi * 16];
      bf_[ni] = cat16(*(const v8bf*)p, *(const v8bf*)(p + 8));
    }
#pragma unroll
    for (int mi = 0; mi < 2; ++mi) {  // A 16x32 frag: lane-half picks K 0-15 / 16-31
      const __bf16* p = &lA[mo + mi * 16 + lo16][hi * 16];
      v16bf af = cat16(*(const v8bf*)p, *(const v8bf*)(p + 8));
#pragma unroll
      for (int ni = 0; ni < 2; ++ni)
        acc[mi][ni] = __builtin_amdgcn_wmma_f32_16x16x32_bf16(
            false, af, false, bf_[ni], (short)0, acc[mi][ni], false, false);
    }
  }

  const float qscale = 0.125f * LOG2E;      // fold 1/sqrt(64) and log2(e) into Q
#pragma unroll
  for (int mi = 0; mi < 2; ++mi)
#pragma unroll
    for (int ni = 0; ni < 2; ++ni) {
      int n  = Nbase + no + ni * 16 + lo16; // model-dim column -> (h,d)
      int h  = n >> 6, d = n & 63;
      int m0 = Mbase + mo + mi * 16 + hi * 8;
      int b  = m0 >> 10;
      int s0 = m0 & 1023;
      if (z == 2) {                         // V transposed: 8 consecutive s -> 16B store
        v8bf st;
#pragma unroll
        for (int j = 0; j < 8; ++j) st[j] = f2bf(acc[mi][ni][j]);
        *(v8bf*)&vtws[(((size_t)b * 16 + h) * 64 + d) * 1024 + s0] = st;
      } else {
        __bf16* dst = (z == 0) ? qws : kws;
        float   sc  = (z == 0) ? qscale : 1.0f;
#pragma unroll
        for (int j = 0; j < 8; ++j)
          dst[(((size_t)b * 16 + h) * 1024 + (s0 + j)) * 64 + d] = f2bf(acc[mi][ni][j] * sc);
      }
    }
}

// -----------------------------------------------------------------------------
// Kernel 2: fused flash-style attention with post-softmax multiplicative link
// mask.  One workgroup per (b, h, 64 q-rows); 4 waves, 16 q-rows each.
// link_mask tiles stream via double-buffered GLOBAL_LOAD_ASYNC_TO_LDS_B128.
// -----------------------------------------------------------------------------
__global__ __launch_bounds__(128) void attn_kernel(
    const __bf16* __restrict__ qws, const __bf16* __restrict__ kws,
    const __bf16* __restrict__ vtws,
    const float* __restrict__ am, const float* __restrict__ link,
    float* __restrict__ out)
{
  const int qt = blockIdx.x;   // 16 q-tiles
  const int h  = blockIdx.y;   // 16 heads
  const int b  = blockIdx.z;   // 4 batch

  const int tid  = threadIdx.x;
  const int lane = tid & 31;
  const int w    = tid >> 5;
  const int hi   = lane >> 4;
  const int lo16 = lane & 15;

  __shared__ __align__(16) float  lmask[1024];        // additive mask * log2e
  __shared__ __align__(16) float  lscore[4][16][32];  // per-wave score staging
  __shared__ __align__(16) __bf16 lP[4][16][32];      // per-wave P staging
  __shared__ __align__(16) float  llink[2][64][32];   // double-buffered link tile
  __shared__ float  lred[4][16];                      // per-row broadcast

#pragma unroll
  for (int i = 0; i < 2; ++i) {
    int c4 = (tid + i * 128) * 4;
    v4f m = *(const v4f*)&am[(size_t)b * 1024 + c4];
#pragma unroll
    for (int j = 0; j < 4; ++j) lmask[c4 + j] = m[j] * LOG2E;
  }

  const size_t bh = (size_t)b * 16 + h;
  const int qrow  = qt * 64 + w * 16 + lo16;
  const float* linkbase = link + ((size_t)b * 1024 + qt * 64) * 1024;
  const int ar  = tid >> 3;          // async copy: row per thread-step
  const int ac4 = (tid & 7) * 4;     // async copy: 4-float column base

  // Q A-fragments held in registers for the whole kernel (16x64 per wave).
  v16bf qf[2];
  {
    const __bf16* qp = &qws[(bh * 1024 + qrow) * 64];
#pragma unroll
    for (int kk = 0; kk < 2; ++kk) {
      const __bf16* p = qp + kk * 32 + hi * 16;
      qf[kk] = cat16(*(const v8bf*)p, *(const v8bf*)(p + 8));
    }
  }

  v8f acc[4];
#pragma unroll
  for (int dt = 0; dt < 4; ++dt)
#pragma unroll
    for (int j = 0; j < 8; ++j) acc[dt][j] = 0.0f;
  float mrow = -1e30f;
  float zrow = 0.0f;

  // prefetch first link tile (async -> LDS, ASYNCcnt)
#pragma unroll
  for (int i = 0; i < 4; ++i) {
    int r = ar + i * 16;
    async_b128((unsigned)(uintptr_t)&llink[0][r][ac4],
               linkbase + (size_t)r * 1024 + ac4);
  }

  for (int kb = 0; kb < 1024; kb += 32) {
    const int p = (kb >> 5) & 1;
    wait_async0();                     // current tile landed in LDS
    __syncthreads();                   // visible to all waves; prev buffer free
    if (kb + 32 < 1024) {              // prefetch next tile into other buffer
#pragma unroll
      for (int i = 0; i < 4; ++i) {
        int r = ar + i * 16;
        async_b128((unsigned)(uintptr_t)&llink[p ^ 1][r][ac4],
                   linkbase + (size_t)r * 1024 + (kb + 32) + ac4);
      }
    }

    // ---- scores: 16x32 tile = 2 N-subtiles x 2 K-steps of WMMA -------------
#pragma unroll
    for (int sub = 0; sub < 2; ++sub) {
      int n = kb + sub * 16 + lo16;                     // key token
      const __bf16* kp = &kws[(bh * 1024 + n) * 64 + hi * 16];
      v8f sc = {0.f, 0.f, 0.f, 0.f, 0.f, 0.f, 0.f, 0.f};
#pragma unroll
      for (int kk = 0; kk < 2; ++kk) {
        const __bf16* pk = kp + kk * 32;
        v16bf bfr = cat16(*(const v8bf*)pk, *(const v8bf*)(pk + 8));
        sc = __builtin_amdgcn_wmma_f32_16x16x32_bf16(
            false, qf[kk], false, bfr, (short)0, sc, false, false);
      }
#pragma unroll
      for (int j = 0; j < 8; ++j)           // D-layout -> LDS
        lscore[w][j + hi * 8][sub * 16 + lo16] = sc[j];
    }

    // ---- online softmax (exp2 domain); lane: row lo16, col half hi ---------
    float vbuf[16];
    float lm = -1e30f;
#pragma unroll
    for (int j = 0; j < 16; ++j) {
      float vv = lscore[w][lo16][hi * 16 + j] + lmask[kb + hi * 16 + j];
      vbuf[j] = vv;
      lm = fmaxf(lm, vv);
    }
    lm = fmaxf(lm, __shfl_xor(lm, 16));
    float mnew = fmaxf(mrow, lm);
    float corr = __builtin_amdgcn_exp2f(mrow - mnew);
    float ps = 0.0f;
#pragma unroll
    for (int j = 0; j < 16; ++j) {
      float pr = __builtin_amdgcn_exp2f(vbuf[j] - mnew);
      ps += pr;                                              // Z: un-linked
      float pl = pr * llink[p][w * 16 + lo16][hi * 16 + j];  // acc: link-weighted
      lP[w][lo16][hi * 16 + j] = f2bf(pl);
    }
    ps += __shfl_xor(ps, 16);
    zrow = zrow * corr + ps;
    mrow = mnew;

    if (hi == 0) lred[w][lo16] = corr;                   // per-row rescale
    float cf[8];
#pragma unroll
    for (int j = 0; j < 8; ++j) cf[j] = lred[w][j + hi * 8];
#pragma unroll
    for (int dt = 0; dt < 4; ++dt)
#pragma unroll
      for (int j = 0; j < 8; ++j) acc[dt][j] *= cf[j];

    // ---- PV: P (16x32 bf16, A-layout from LDS) @ V (32x64) -----------------
    const __bf16* pp = &lP[w][lo16][hi * 16];
    v16bf pf = cat16(*(const v8bf*)pp, *(const v8bf*)(pp + 8));
#pragma unroll
    for (int dt = 0; dt < 4; ++dt) {
      int d = dt * 16 + lo16;
      const __bf16* vp = &vtws[(bh * 64 + d) * 1024 + kb + hi * 16];
      v16bf vf = cat16(*(const v8bf*)vp, *(const v8bf*)(vp + 8));
      acc[dt] = __builtin_amdgcn_wmma_f32_16x16x32_bf16(
          false, pf, false, vf, (short)0, acc[dt], false, false);
    }
  }

  // ---- finalize: divide by Z, store [B,S,DM] f32 ---------------------------
  if (hi == 0) lred[w][lo16] = 1.0f / zrow;
  float zf[8];
#pragma unroll
  for (int j = 0; j < 8; ++j) zf[j] = lred[w][j + hi * 8];
#pragma unroll
  for (int dt = 0; dt < 4; ++dt) {
    int d = dt * 16 + lo16;
#pragma unroll
    for (int j = 0; j < 8; ++j) {
      int srow = qt * 64 + w * 16 + hi * 8 + j;
      out[((size_t)b * 1024 + srow) * 1024 + h * 64 + d] = acc[dt][j] * zf[j];
    }
  }
}

// -----------------------------------------------------------------------------
extern "C" void kernel_launch(void* const* d_in, const int* in_sizes, int n_in,
                              void* d_out, int out_size, void* d_ws, size_t ws_size,
                              hipStream_t stream) {
  const float* hs   = (const float*)d_in[0];
  const float* am   = (const float*)d_in[1];
  const float* link = (const float*)d_in[2];
  const float* Wq   = (const float*)d_in[3];
  const float* bq   = (const float*)d_in[4];
  const float* Wk   = (const float*)d_in[5];
  const float* bk   = (const float*)d_in[6];
  const float* Wv   = (const float*)d_in[7];
  const float* bv   = (const float*)d_in[8];

  const size_t QN = (size_t)4 * 16 * 1024 * 64;  // elements per Q/K/V (bf16)
  __bf16* qws  = (__bf16*)d_ws;
  __bf16* kws  = qws + QN;
  __bf16* vtws = kws + QN;

  qkv_kernel<<<dim3(64, 16, 3), 128, 0, stream>>>(hs, Wq, bq, Wk, bk, Wv, bv,
                                                  qws, kws, vtws);
  attn_kernel<<<dim3(16, 16, 4), 128, 0, stream>>>(qws, kws, vtws, am, link,
                                                   (float*)d_out);
}
